// GNN_41205916238177
// MI455X (gfx1250) — matrix operand order repaired and checked by reading, not compiled, not run on hardware
//
#include <hip/hip_runtime.h>
#include <hip/hip_bf16.h>

// ---------------------------------------------------------------------------
// MI455X (gfx1250) implementation.
//  - wave32, WMMA 16x16x32 f16->f32 for the channel-mix GEMMs and final dense
//  - LDS staging of weights/activation tiles (layouts chosen so fragment
//    gathers are contiguous -> ds_load_b128), global_prefetch for next K tile
//  - warp/mask stage is a fused VALU gather kernel (memory bound by design)
// ---------------------------------------------------------------------------

typedef __attribute__((ext_vector_type(16))) _Float16 v16h;
typedef __attribute__((ext_vector_type(8)))  float    v8f;

constexpr int BATCH = 16;
constexpr int CHAN  = 128;
constexpr int INCH  = 64;
constexpr int HDIM  = 96;
constexpr int HW    = HDIM * HDIM;          // 9216
constexpr int NOUT  = 1000;
constexpr int NTILE = 64;                   // pixels per block tile
constexpr int XPITCH = 40;                  // halves per sX row (80B, bank-spread)
constexpr long FEATN = (long)BATCH * CHAN * HW; // 18,874,368

// ---------------------------------------------------------------------------
// Channel-mix GEMM: OUT[b, m, p] = sum_k LIN[m,k] * X[b,k,p]
// M = 128 (8 waves x 16), N-tile = 64 pixels, K stepped by 32 (one WMMA).
// A-frag layout per ISA 7.12.2 (16-bit A 16x32): lane(0..15)=M row,
//   elems 0..7 -> K = 8*hi + e, elems 8..15 -> K = 16 + 8*hi + (e-8)
//   => two contiguous 8-half runs starting at K = 8*hi and K = 16 + 8*hi.
// B-frag (32x16 K-major): lane&15 = N col, same K order as A
//   => two contiguous 8-half runs per lane in the [pixel][k] staged tile.
// ---------------------------------------------------------------------------
template <int CIN>
__global__ __launch_bounds__(256) void mix_wmma(
    const float* __restrict__ X,    // [BATCH, CIN, HW]
    const float* __restrict__ LIN,  // [CHAN, CIN]
    float* __restrict__ OUT)        // [BATCH, CHAN, HW]
{
  __shared__ _Float16 sLin[CHAN * CIN];           // 16 or 32 KB
  __shared__ _Float16 sX[NTILE * XPITCH];         // [pixel][k], 5 KB

  const int tid    = threadIdx.x;
  const int wave   = tid >> 5;
  const int lane   = tid & 31;
  const int ln     = lane & 15;
  const int hi     = lane >> 4;
  const int m_base = wave * 16;

  const int blk    = blockIdx.x;                  // BATCH * (HW/NTILE)
  const int b      = blk / (HW / NTILE);
  const int n_base = (blk % (HW / NTILE)) * NTILE;

  // Stage mixing weights once, converted to f16 (row-major [m][k]).
  for (int i = tid; i < CHAN * CIN; i += 256) sLin[i] = (_Float16)LIN[i];

  const v8f zero8 = {0.f, 0.f, 0.f, 0.f, 0.f, 0.f, 0.f, 0.f};
  v8f acc[4];
#pragma unroll
  for (int i = 0; i < 4; ++i) acc[i] = zero8;

  for (int k0 = 0; k0 < CIN; k0 += 32) {
    __syncthreads();
    // Cooperative load of the 32(K) x 64(N) activation tile, transposed into
    // [pixel][k] f16 rows so B-fragment gathers are contiguous.
    for (int i = tid; i < 32 * NTILE; i += 256) {
      const int kk = i >> 6;                      // 0..31
      const int nn = i & (NTILE - 1);             // 0..63
      sX[nn * XPITCH + kk] =
          (_Float16)X[(size_t)(b * CIN + k0 + kk) * HW + n_base + nn];
    }
    // Prefetch row starts of the next K tile (emits global_prefetch_b8).
    if (k0 + 32 < CIN && tid < 32) {
      __builtin_prefetch(&X[(size_t)(b * CIN + k0 + 32 + tid) * HW + n_base], 0, 1);
    }
    __syncthreads();

    // A fragment: two contiguous 8-half runs -> ds_load_b128 x2.
    v16h a;
    {
      const _Float16* ar = &sLin[(m_base + ln) * CIN + k0];
#pragma unroll
      for (int e = 0; e < 8; ++e) a[e] = ar[8 * hi + e];
#pragma unroll
      for (int e = 0; e < 8; ++e) a[8 + e] = ar[16 + 8 * hi + e];
    }

#pragma unroll
    for (int ns = 0; ns < 4; ++ns) {
      // B fragment: two contiguous 8-half runs per lane -> ds_load_b128 x2.
      v16h bb;
      const _Float16* br = &sX[(ns * 16 + ln) * XPITCH + 8 * hi];
#pragma unroll
      for (int e = 0; e < 8; ++e) { bb[e] = br[e]; bb[8 + e] = br[16 + e]; }
      acc[ns] = __builtin_amdgcn_wmma_f32_16x16x32_f16(
          /*neg_a=*/false, a, /*neg_b=*/false, bb,
          /*c_mod=*/(short)0, acc[ns], /*reuse_a=*/false, /*reuse_b=*/false);
    }
  }

  // D layout: VGPR r -> M = m_base + r + 8*hi, N = lane&15.
#pragma unroll
  for (int ns = 0; ns < 4; ++ns) {
#pragma unroll
    for (int r = 0; r < 8; ++r) {
      const int m = m_base + r + 8 * hi;
      const int n = n_base + ns * 16 + ln;
      OUT[(size_t)(b * CHAN + m) * HW + n] = acc[ns][r];
    }
  }
}

// ---------------------------------------------------------------------------
// Fused affine-grid bilinear sample (zeros padding) * box mask (+ residual).
// mask(c,h,w) = bilinear-sample of ones with `box` affine = sum of in-bounds
// tap weights -> computed analytically, no memory reads.
// ---------------------------------------------------------------------------
__device__ __forceinline__ float tap_val(const float* __restrict__ img,
                                         int yi, int xi, float w) {
  const bool valid = (xi >= 0) & (xi < HDIM) & (yi >= 0) & (yi < HDIM);
  const int xc = xi < 0 ? 0 : (xi > HDIM - 1 ? HDIM - 1 : xi);
  const int yc = yi < 0 ? 0 : (yi > HDIM - 1 ? HDIM - 1 : yi);
  return valid ? img[yc * HDIM + xc] * w : 0.f;
}

__device__ __forceinline__ float tap_msk(int yi, int xi, float w) {
  return ((xi >= 0) & (xi < HDIM) & (yi >= 0) & (yi < HDIM)) ? w : 0.f;
}

__global__ __launch_bounds__(256) void warp_geom(
    const float* __restrict__ MIXED,  // [BATCH, CHAN, HW]
    const float* __restrict__ GEO,    // [CHAN, 2, 3]
    const float* __restrict__ BOX,    // [CHAN, 2, 3]
    const float* __restrict__ RESID,  // [BATCH, CHAN, HW] or nullptr
    float* __restrict__ OUT)          // [BATCH, CHAN, HW]
{
  const size_t idx = (size_t)blockIdx.x * 256 + threadIdx.x; // < FEATN
  const int p  = (int)(idx % HW);
  const int c  = (int)((idx / HW) % CHAN);
  const int h  = p / HDIM;
  const int w  = p % HDIM;

  const float xs = (w + 0.5f) * (2.0f / HDIM) - 1.0f;
  const float ys = (h + 0.5f) * (2.0f / HDIM) - 1.0f;

  // geo sample
  const float* g  = GEO + c * 6;
  const float gx  = g[0] * xs + g[1] * ys + g[2];
  const float gy  = g[3] * xs + g[4] * ys + g[5];
  const float ix  = ((gx + 1.f) * (float)HDIM - 1.f) * 0.5f;
  const float iy  = ((gy + 1.f) * (float)HDIM - 1.f) * 0.5f;
  const float x0f = floorf(ix), y0f = floorf(iy);
  const int   x0  = (int)x0f,   y0  = (int)y0f;
  const float wx1 = ix - x0f,   wy1 = iy - y0f;
  const float wx0 = 1.f - wx1,  wy0 = 1.f - wy1;

  const float* row = MIXED + (idx - p);  // base of [b,c] plane
  float val = tap_val(row, y0,     x0,     wy0 * wx0)
            + tap_val(row, y0,     x0 + 1, wy0 * wx1)
            + tap_val(row, y0 + 1, x0,     wy1 * wx0)
            + tap_val(row, y0 + 1, x0 + 1, wy1 * wx1);

  // box mask (sample of ones)
  const float* q  = BOX + c * 6;
  const float bx  = q[0] * xs + q[1] * ys + q[2];
  const float by  = q[3] * xs + q[4] * ys + q[5];
  const float jx  = ((bx + 1.f) * (float)HDIM - 1.f) * 0.5f;
  const float jy  = ((by + 1.f) * (float)HDIM - 1.f) * 0.5f;
  const float u0f = floorf(jx), v0f = floorf(jy);
  const int   u0  = (int)u0f,   v0  = (int)v0f;
  const float ux1 = jx - u0f,   vy1 = jy - v0f;
  const float ux0 = 1.f - ux1,  vy0 = 1.f - vy1;
  const float mask = tap_msk(v0,     u0,     vy0 * ux0)
                   + tap_msk(v0,     u0 + 1, vy0 * ux1)
                   + tap_msk(v0 + 1, u0,     vy1 * ux0)
                   + tap_msk(v0 + 1, u0 + 1, vy1 * ux1);

  float out = mask * val;
  if (RESID) out += RESID[idx];
  OUT[idx] = out;
}

// ---------------------------------------------------------------------------
// Mean pool over HW: one block per (b,c).
// ---------------------------------------------------------------------------
__global__ __launch_bounds__(256) void pool_mean(
    const float* __restrict__ FEAT, float* __restrict__ POOLED)
{
  __shared__ float red[256];
  const int bc = blockIdx.x;                    // 0 .. BATCH*CHAN-1
  const float* src = FEAT + (size_t)bc * HW;
  float s = 0.f;
  for (int i = threadIdx.x; i < HW; i += 256) s += src[i];
  red[threadIdx.x] = s;
  __syncthreads();
  for (int off = 128; off > 0; off >>= 1) {
    if (threadIdx.x < off) red[threadIdx.x] += red[threadIdx.x + off];
    __syncthreads();
  }
  if (threadIdx.x == 0) POOLED[bc] = red[0] * (1.0f / (float)HW);
}

// ---------------------------------------------------------------------------
// logits[m,n] = sum_k pooled[m,k] * dense_w[n,k] + bias[n]
// M=16 (batch), K=128 (4 WMMA steps), one wave per 16-column tile (63 tiles).
// Tail handling: clamp the weight-row index so every lane loads a valid,
// pair of contiguous 32B runs per K step (global_load_b128); garbage columns n>=1000 are simply
// never stored. This keeps EXEC uniform through the WMMA and avoids the
// per-element predicated load chains seen in the previous build.
// ---------------------------------------------------------------------------
__global__ __launch_bounds__(256) void dense_wmma(
    const float* __restrict__ POOLED,   // [16, 128]
    const float* __restrict__ DW,       // [1000, 128]
    const float* __restrict__ DB,       // [1000]
    float* __restrict__ LOGITS)         // [16, 1000]
{
  const int wave = (int)((blockIdx.x * 256 + threadIdx.x) >> 5);
  const int lane = threadIdx.x & 31;
  if (wave >= (NOUT + 15) / 16) return;         // uniform per-wave guard
  const int ln = lane & 15, hi = lane >> 4;
  const int n  = wave * 16 + ln;
  const int nc = n < (NOUT - 1) ? n : (NOUT - 1);   // clamped row for loads
  const float* wrow = DW + (size_t)nc * CHAN;
  const float* prow = POOLED + ln * CHAN;

  const v8f zero8 = {0.f, 0.f, 0.f, 0.f, 0.f, 0.f, 0.f, 0.f};
  v8f acc = zero8;
#pragma unroll
  for (int k0 = 0; k0 < CHAN; k0 += 32) {
    v16h a, bb;
#pragma unroll
    for (int e = 0; e < 8; ++e) a[e] = (_Float16)prow[k0 + 8 * hi + e];
#pragma unroll
    for (int e = 0; e < 8; ++e) a[8 + e] = (_Float16)prow[k0 + 16 + 8 * hi + e];
#pragma unroll
    for (int e = 0; e < 8; ++e) { bb[e] = (_Float16)wrow[k0 + 8 * hi + e]; bb[8 + e] = (_Float16)wrow[k0 + 16 + 8 * hi + e]; }
    acc = __builtin_amdgcn_wmma_f32_16x16x32_f16(
        false, a, false, bb, (short)0, acc, false, false);
  }

  if (n < NOUT) {
    const float bias = DB[n];
#pragma unroll
    for (int r = 0; r < 8; ++r) {
      const int m = r + 8 * hi;                 // batch row
      LOGITS[m * NOUT + n] = acc[r] + bias;
    }
  }
}

// ---------------------------------------------------------------------------
extern "C" void kernel_launch(void* const* d_in, const int* in_sizes, int n_in,
                              void* d_out, int out_size, void* d_ws, size_t ws_size,
                              hipStream_t stream) {
  const float* x       = (const float*)d_in[0];   // [16,64,96,96]
  const float* in_geo  = (const float*)d_in[1];   // [128,2,3]
  const float* in_box  = (const float*)d_in[2];
  const float* in_lin  = (const float*)d_in[3];   // [128,64]
  const float* lay_geo = (const float*)d_in[4];   // [4,128,2,3]
  const float* lay_box = (const float*)d_in[5];
  const float* lay_lin = (const float*)d_in[6];   // [4,128,128]
  const float* dense_w = (const float*)d_in[7];   // [1000,128]
  const float* dense_b = (const float*)d_in[8];   // [1000]

  float* out    = (float*)d_out;
  float* logits = out;                    // 16*1000
  float* feat   = out + BATCH * NOUT;     // 16*128*96*96

  // Workspace: two feature ping-pong buffers + pooled (needs ~151 MB).
  float* cur    = (float*)d_ws;
  float* mixed  = cur + FEATN;
  float* pooled = mixed + FEATN;          // 2048 floats

  const dim3 blk(256);
  const dim3 mix_grid(BATCH * (HW / NTILE));      // 2304 blocks
  const dim3 warp_grid((unsigned)(FEATN / 256));  // 73728 blocks

  // Layer 0: K = 64, no residual.
  mix_wmma<INCH><<<mix_grid, blk, 0, stream>>>(x, in_lin, mixed);
  warp_geom<<<warp_grid, blk, 0, stream>>>(mixed, in_geo, in_box, nullptr, cur);

  // Layers 1..4: K = 128, residual; final layer writes d_out feature region.
  for (int i = 0; i < 4; ++i) {
    const float* lin = lay_lin + (size_t)i * CHAN * CHAN;
    const float* geo = lay_geo + (size_t)i * CHAN * 6;
    const float* box = lay_box + (size_t)i * CHAN * 6;
    mix_wmma<CHAN><<<mix_grid, blk, 0, stream>>>(cur, lin, mixed);
    float* dst = (i == 3) ? feat : cur;
    warp_geom<<<warp_grid, blk, 0, stream>>>(mixed, geo, box, cur, dst);
  }

  pool_mean<<<dim3(BATCH * CHAN), blk, 0, stream>>>(feat, pooled);
  dense_wmma<<<dim3(8), blk, 0, stream>>>(pooled, dense_w, dense_b, logits);
}
